// GCNDeno_60687887892590
// MI455X (gfx1250) — matrix operand order repaired and checked by reading, not compiled, not run on hardware
//
#include <hip/hip_runtime.h>

typedef __attribute__((ext_vector_type(16))) _Float16 v16h;
typedef __attribute__((ext_vector_type(8)))  _Float16 v8h;
typedef __attribute__((ext_vector_type(8)))  float    v8f;

#define BB   128
#define NN   512
#define DD   128
#define RR   512
#define OUTC (DD + RR)   // 640

// ---------------------------------------------------------------------------
// WMMA helpers (A/B k order contiguous, k = 16*(lane/16) + s, not the ISA table order; wave32)
// ---------------------------------------------------------------------------
__device__ __forceinline__ v8f wmma_f16(v16h a, v16h b, v8f c) {
  // D = A(16x32,f16) * B(32x16,f16) + C(16x16,f32)
  return __builtin_amdgcn_wmma_f32_16x16x32_f16(
      /*neg_a=*/false, a, /*neg_b=*/false, b,
      /*c_mod=*/(short)0, c, /*reuse_a=*/false, /*reuse_b=*/false);
}

// A fragment: 16x32 tile, row-major source. `A` points at (row0, k0).
__device__ __forceinline__ v16h load_a_rowmajor(const _Float16* A, int lda, int lane) {
  int m  = lane & 15;
  int kb = (lane >> 4) << 4;            // 0 or 16
  const _Float16* p = A + (long)m * lda + kb;
  v16h r;
#pragma unroll
  for (int i = 0; i < 8; ++i) r[i] = p[i];
#pragma unroll
  for (int i = 0; i < 8; ++i) r[i + 8] = p[8 + i];
  return r;
}

// B fragment where logical B(k,n) = M[n*ld + k] (M is the transposed/N-major
// store of B). `M` points at (col0, k0). 32 contiguous bytes per lane.
__device__ __forceinline__ v16h load_b_transposed(const _Float16* M, int ld, int lane) {
  int n  = lane & 15;
  int kb = (lane >> 4) << 4;
  const _Float16* p = M + (long)n * ld + kb;
  v16h r;
#pragma unroll
  for (int i = 0; i < 16; ++i) r[i] = p[i];
  return r;
}

// ---------------------------------------------------------------------------
// K0: weights f32 -> f16, transposed per Chebyshev order: WT[t][e][d]=W[t][d][e]
// ---------------------------------------------------------------------------
__global__ void k_convert_w_t(const float* __restrict__ src,
                              _Float16* __restrict__ dst) {
  int i = blockIdx.x * blockDim.x + threadIdx.x;     // over 3*DD*DD
  int t = i / (DD * DD);
  int r = i - t * (DD * DD);
  int d = r >> 7, e = r & (DD - 1);
  dst[t * DD * DD + e * DD + d] = (_Float16)src[i];
}

// ---------------------------------------------------------------------------
// K1: gather app embeddings, write raw f32 to output cols [D, D+R),
//     write L2-normalized f16 rows for the similarity GEMM.
// ---------------------------------------------------------------------------
__global__ void k_gather_norm_app(const int* __restrict__ recd,
                                  const float* __restrict__ table,
                                  float* __restrict__ out,
                                  _Float16* __restrict__ appf) {
  int row = blockIdx.x;                       // 0 .. B*N-1
  int tok = recd[row];
  const float* src = table + (long)tok * RR;
  __shared__ float red[256];
  float vals[2];
  float ss = 0.f;
  int c = 0;
  for (int j = threadIdx.x; j < RR; j += 256, ++c) {
    float v = src[j];
    vals[c] = v;
    ss += v * v;
  }
  red[threadIdx.x] = ss;
  __syncthreads();
  for (int s = 128; s > 0; s >>= 1) {
    if (threadIdx.x < s) red[threadIdx.x] += red[threadIdx.x + s];
    __syncthreads();
  }
  float rn = 1.0f / fmaxf(sqrtf(red[0]), 1e-12f);
  float*    orow = out  + (long)row * OUTC + DD;
  _Float16* arow = appf + (long)row * RR;
  c = 0;
  for (int j = threadIdx.x; j < RR; j += 256, ++c) {
    orow[j] = vals[c];
    arow[j] = (_Float16)(vals[c] * rn);
  }
}

// ---------------------------------------------------------------------------
// K2: per-batch similarity GEMM  A[b] = F[b] * F[b]^T   (512x512, K=512)
//     8 waves / block, wave tile 32x64 (2x4 WMMA accums), block tile 128x128.
// ---------------------------------------------------------------------------
__global__ void k_sim_gemm(const _Float16* __restrict__ appf,
                           float* __restrict__ Aout) {
  int b    = blockIdx.x;
  int wave = threadIdx.x >> 5, lane = threadIdx.x & 31;
  int row0 = blockIdx.y * 128 + (wave >> 1) * 32;
  int col0 = blockIdx.z * 128 + (wave & 1) * 64;
  const _Float16* X = appf + (long)b * NN * RR;
  v8f acc[2][4] = {};
  for (int k = 0; k < RR; k += 32) {
    v16h a0 = load_a_rowmajor(X + (long)row0 * RR + k, RR, lane);
    v16h a1 = load_a_rowmajor(X + (long)(row0 + 16) * RR + k, RR, lane);
    v16h bf[4];
#pragma unroll
    for (int j = 0; j < 4; ++j)
      bf[j] = load_b_transposed(X + (long)(col0 + j * 16) * RR + k, RR, lane);
#pragma unroll
    for (int j = 0; j < 4; ++j) {
      acc[0][j] = wmma_f16(a0, bf[j], acc[0][j]);
      acc[1][j] = wmma_f16(a1, bf[j], acc[1][j]);
    }
  }
  float* C = Aout + (long)b * NN * NN;
  int n = lane & 15, mb = (lane >> 4) * 8;
#pragma unroll
  for (int i = 0; i < 2; ++i)
#pragma unroll
    for (int j = 0; j < 4; ++j)
#pragma unroll
      for (int v = 0; v < 8; ++v)
        C[(long)(row0 + i * 16 + mb + v) * NN + col0 + j * 16 + n] = acc[i][j][v];
}

// ---------------------------------------------------------------------------
// K3: per-batch mean of strict-upper-triangle (sum / count-nonzero)
// ---------------------------------------------------------------------------
__global__ void k_mean(const float* __restrict__ A, float* __restrict__ meanb) {
  int b = blockIdx.x;
  const float* Ab = A + (long)b * NN * NN;
  __shared__ float ssum[256];
  __shared__ float scnt[256];
  float s = 0.f, c = 0.f;
  for (int idx = threadIdx.x; idx < NN * NN; idx += 256) {
    int i = idx >> 9, j = idx & (NN - 1);
    if (j > i) {
      float v = Ab[idx];
      s += v;
      if (v != 0.0f) c += 1.0f;
    }
  }
  ssum[threadIdx.x] = s;
  scnt[threadIdx.x] = c;
  __syncthreads();
  for (int st = 128; st > 0; st >>= 1) {
    if (threadIdx.x < st) {
      ssum[threadIdx.x] += ssum[threadIdx.x + st];
      scnt[threadIdx.x] += scnt[threadIdx.x + st];
    }
    __syncthreads();
  }
  if (threadIdx.x == 0) meanb[b] = ssum[0] / fmaxf(scnt[0], 1.0f);
}

// ---------------------------------------------------------------------------
// K4: per-row degree -> dinv = deg>0 ? rsqrt(deg) : 0. One block per row.
// ---------------------------------------------------------------------------
__global__ void k_deg(const float* __restrict__ A,
                      const float* __restrict__ meanb,
                      float* __restrict__ dinv) {
  int row = blockIdx.x;                // b*N + i
  int b = row >> 9, i = row & (NN - 1);
  float mean = meanb[b];
  const float* Ar = A + (long)row * NN;
  __shared__ float red[256];
  float c = 0.f;
  for (int j = threadIdx.x; j < NN; j += 256) {
    float v = Ar[j];
    if (j != i && v >= mean && v != 0.0f) c += 1.0f;
  }
  red[threadIdx.x] = c;
  __syncthreads();
  for (int s = 128; s > 0; s >>= 1) {
    if (threadIdx.x < s) red[threadIdx.x] += red[threadIdx.x + s];
    __syncthreads();
  }
  if (threadIdx.x == 0) dinv[row] = red[0] > 0.f ? rsqrtf(red[0]) : 0.f;
}

// ---------------------------------------------------------------------------
// K5: build S (f16):  S_ij = edge ? -dinv_i * dinv_j : 0
// ---------------------------------------------------------------------------
__global__ void k_build_s(const float* __restrict__ A,
                          const float* __restrict__ meanb,
                          const float* __restrict__ dinv,
                          _Float16* __restrict__ S) {
  long total = (long)BB * NN * NN;
  for (long idx = (long)blockIdx.x * blockDim.x + threadIdx.x; idx < total;
       idx += (long)gridDim.x * blockDim.x) {
    int  j  = (int)(idx & (NN - 1));
    long ri = idx >> 9;                       // b*N + i
    int  i  = (int)(ri & (NN - 1));
    int  b  = (int)(ri >> 9);
    float v = A[idx];
    float s = 0.f;
    if (j != i && v >= meanb[b] && v != 0.0f)
      s = -dinv[ri] * dinv[(long)b * NN + j];
    S[idx] = (_Float16)s;
  }
}

// ---------------------------------------------------------------------------
// K6: gather node embeddings -> f32 row-major + f16 row-major + f16 transposed
// ---------------------------------------------------------------------------
__global__ void k_gather_x(const int* __restrict__ seq,
                           const float* __restrict__ table,
                           float* __restrict__ xf,
                           _Float16* __restrict__ xh,
                           _Float16* __restrict__ xhT) {
  long idx = (long)blockIdx.x * blockDim.x + threadIdx.x;   // B*N*D
  int  d   = (int)(idx & (DD - 1));
  long row = idx >> 7;                                      // b*N + n
  int  b   = (int)(row >> 9);
  int  n   = (int)(row & (NN - 1));
  int  tok = seq[row];
  float v = table[(long)tok * DD + d];
  xf[idx] = v;
  xh[idx] = (_Float16)v;
  xhT[((long)b * DD + d) * NN + n] = (_Float16)v;
}

// ---------------------------------------------------------------------------
// K7: per-batch GEMM  C = S[b] @ X[b]   (512x128, K=512); X given transposed.
//     Wave tile 32x64, block tile 128x128 (full output width).
//     mode 0: outh = C  (row-major) and outhT = C^T (coalesced v8h stores)
//     mode 1: outh = 2*C - T0   (row-major only; Chebyshev tx2)
// ---------------------------------------------------------------------------
__global__ void k_gemm_sx(const _Float16* __restrict__ S,
                          const _Float16* __restrict__ XT,
                          const float* __restrict__ T0,
                          _Float16* __restrict__ outh,
                          _Float16* __restrict__ outhT, int mode) {
  int b    = blockIdx.x;
  int wave = threadIdx.x >> 5, lane = threadIdx.x & 31;
  int row0 = blockIdx.y * 128 + (wave >> 1) * 32;
  int col0 = (wave & 1) * 64;
  const _Float16* Sb  = S  + (long)b * NN * NN;
  const _Float16* XTb = XT + (long)b * DD * NN;
  v8f acc[2][4] = {};
  for (int k = 0; k < NN; k += 32) {
    v16h a0 = load_a_rowmajor(Sb + (long)row0 * NN + k, NN, lane);
    v16h a1 = load_a_rowmajor(Sb + (long)(row0 + 16) * NN + k, NN, lane);
    v16h bf[4];
#pragma unroll
    for (int j = 0; j < 4; ++j)
      bf[j] = load_b_transposed(XTb + (long)(col0 + j * 16) * NN + k, NN, lane);
#pragma unroll
    for (int j = 0; j < 4; ++j) {
      acc[0][j] = wmma_f16(a0, bf[j], acc[0][j]);
      acc[1][j] = wmma_f16(a1, bf[j], acc[1][j]);
    }
  }
  int  n = lane & 15, mb = (lane >> 4) * 8;
  long base  = (long)b * NN * DD;
  long tbase = (long)b * DD * NN;
#pragma unroll
  for (int i = 0; i < 2; ++i)
#pragma unroll
    for (int j = 0; j < 4; ++j) {
      int c     = col0 + j * 16 + n;
      int rbase = row0 + i * 16 + mb;
      v8h pack;
#pragma unroll
      for (int v = 0; v < 8; ++v) {
        long r    = rbase + v;
        long cidx = base + r * DD + c;
        float val = acc[i][j][v];
        if (mode == 1) val = 2.0f * val - T0[cidx];
        _Float16 h = (_Float16)val;
        outh[cidx] = h;
        pack[v] = h;
      }
      if (mode == 0)
        *(v8h*)(outhT + tbase + (long)c * NN + rbase) = pack;
    }
}

// ---------------------------------------------------------------------------
// K8: batched node-feature GEMM over all B*N rows:
//     Y = relu(T0@W0 + T1@W1 + T2@W2 + bias); weights given transposed.
//     Wave tile 32x64, block tile 128x128 (full output width).
//     mode 0: write outf (f32, ld=DD), outh (f16, ld=DD), outhT ([b][DD][NN])
//     mode 1: write outf with leading dim `oldc` (final output, ld=640)
// ---------------------------------------------------------------------------
__global__ void k_gemm_w(const _Float16* __restrict__ T0,
                         const _Float16* __restrict__ T1,
                         const _Float16* __restrict__ T2,
                         const _Float16* __restrict__ WT,
                         const float* __restrict__ bias,
                         float* __restrict__ outf, _Float16* __restrict__ outh,
                         _Float16* __restrict__ outhT, int oldc, int mode) {
  int wave = threadIdx.x >> 5, lane = threadIdx.x & 31;
  int row0 = blockIdx.x * 128 + (wave >> 1) * 32;
  int col0 = (wave & 1) * 64;
  v8f acc[2][4] = {};
  const _Float16* Ts[3] = {T0, T1, T2};
#pragma unroll
  for (int t = 0; t < 3; ++t) {
    const _Float16* Tm = Ts[t];
    const _Float16* W  = WT + t * DD * DD;        // W^T: [e][d]
    for (int k = 0; k < DD; k += 32) {
      v16h a0 = load_a_rowmajor(Tm + (long)row0 * DD + k, DD, lane);
      v16h a1 = load_a_rowmajor(Tm + (long)(row0 + 16) * DD + k, DD, lane);
      v16h bf[4];
#pragma unroll
      for (int j = 0; j < 4; ++j)
        bf[j] = load_b_transposed(W + (long)(col0 + j * 16) * DD + k, DD, lane);
#pragma unroll
      for (int j = 0; j < 4; ++j) {
        acc[0][j] = wmma_f16(a0, bf[j], acc[0][j]);
        acc[1][j] = wmma_f16(a1, bf[j], acc[1][j]);
      }
    }
  }
  int n = lane & 15, mb = (lane >> 4) * 8;
#pragma unroll
  for (int i = 0; i < 2; ++i)
#pragma unroll
    for (int j = 0; j < 4; ++j) {
      int c     = col0 + j * 16 + n;
      int rbase = row0 + i * 16 + mb;
      v8h pack;
#pragma unroll
      for (int v = 0; v < 8; ++v) {
        long r = rbase + v;
        float val = fmaxf(acc[i][j][v] + bias[c], 0.0f);
        if (mode == 0) {
          outf[r * DD + c] = val;
          _Float16 h = (_Float16)val;
          outh[r * DD + c] = h;
          pack[v] = h;
        } else {
          outf[r * (long)oldc + c] = val;
        }
      }
      if (mode == 0) {
        int b  = rbase >> 9;                  // 128-row block stays in one batch
        int nl = rbase & (NN - 1);
        *(v8h*)(outhT + ((long)b * DD + c) * NN + nl) = pack;
      }
    }
}

// ---------------------------------------------------------------------------
// Host launch
// ---------------------------------------------------------------------------
extern "C" void kernel_launch(void* const* d_in, const int* in_sizes, int n_in,
                              void* d_out, int out_size, void* d_ws, size_t ws_size,
                              hipStream_t stream) {
  const int*   input_seq = (const int*)d_in[0];
  const int*   recd_tok  = (const int*)d_in[1];
  const float* embed_tab = (const float*)d_in[2];
  const float* rec_tab   = (const float*)d_in[3];
  const float* cheb_w1   = (const float*)d_in[4];
  const float* cheb_b1   = (const float*)d_in[5];
  const float* cheb_w2   = (const float*)d_in[6];
  const float* cheb_b2   = (const float*)d_in[7];
  float* out = (float*)d_out;

  // ---- workspace carve-up (~269 MB total) ----
  char* ws = (char*)d_ws;
  size_t off = 0;
  _Float16* appf = (_Float16*)(ws + off); off += (size_t)BB * NN * RR * 2;  // 64 MB
  float*    Abuf = (float*)   (ws + off); off += (size_t)BB * NN * NN * 4;  // 128 MB
  _Float16* Sh   = (_Float16*)(ws + off); off += (size_t)BB * NN * NN * 2;  // 64 MB
  float*    meanb= (float*)   (ws + off); off += 4096;
  float*    dinv = (float*)   (ws + off); off += (size_t)BB * NN * 4;
  _Float16* w1hT = (_Float16*)(ws + off); off += (size_t)3 * DD * DD * 2;
  _Float16* w2hT = (_Float16*)(ws + off); off += (size_t)3 * DD * DD * 2;
  // Aliased on top of Abuf: used only AFTER k_build_s (stream-ordered).
  char* rb = (char*)Abuf;
  float*    xf  = (float*)rb;    rb += (size_t)BB * NN * DD * 4;  // 32 MB
  _Float16* xh  = (_Float16*)rb; rb += (size_t)BB * NN * DD * 2;  // 16 MB
  _Float16* t1h = (_Float16*)rb; rb += (size_t)BB * NN * DD * 2;  // 16 MB
  _Float16* t2h = (_Float16*)rb; rb += (size_t)BB * NN * DD * 2;  // 16 MB
  float*    y1f = (float*)rb;    rb += (size_t)BB * NN * DD * 4;  // 32 MB
  _Float16* y1h = (_Float16*)rb; rb += (size_t)BB * NN * DD * 2;  // 16 MB
  // Aliased on top of appf: used only AFTER k_sim_gemm (stream-ordered).
  char* tb = (char*)appf;
  _Float16* xhT  = (_Float16*)tb; tb += (size_t)BB * DD * NN * 2; // 16 MB
  _Float16* t1hT = (_Float16*)tb; tb += (size_t)BB * DD * NN * 2; // 16 MB
  _Float16* y1hT = (_Float16*)tb; tb += (size_t)BB * DD * NN * 2; // 16 MB

  // ---- 0. weights -> f16 transposed ----
  k_convert_w_t<<<(3 * DD * DD + 255) / 256, 256, 0, stream>>>(cheb_w1, w1hT);
  k_convert_w_t<<<(3 * DD * DD + 255) / 256, 256, 0, stream>>>(cheb_w2, w2hT);

  // ---- 1. gather app embeddings: raw -> out[:, D:], normalized f16 -> appf ----
  k_gather_norm_app<<<BB * NN, 256, 0, stream>>>(recd_tok, rec_tab, out, appf);

  // ---- 2. cosine similarity A = F F^T per graph ----
  k_sim_gemm<<<dim3(BB, 4, 4), 256, 0, stream>>>(appf, Abuf);

  // ---- 3. mean of strict upper triangle per graph ----
  k_mean<<<BB, 256, 0, stream>>>(Abuf, meanb);

  // ---- 4. degrees -> dinv ----
  k_deg<<<BB * NN, 256, 0, stream>>>(Abuf, meanb, dinv);

  // ---- 5. S = -(dinv adj dinv), f16 ----
  k_build_s<<<32768, 256, 0, stream>>>(Abuf, meanb, dinv, Sh);

  // ---- 6. gather node embeddings (aliases A/appf regions; stream-ordered) ----
  k_gather_x<<<(BB * NN * DD) / 256, 256, 0, stream>>>(input_seq, embed_tab,
                                                       xf, xh, xhT);

  // ---- layer 1 ----
  k_gemm_sx<<<dim3(BB, 4), 256, 0, stream>>>(Sh, xhT,  nullptr, t1h, t1hT, 0);
  k_gemm_sx<<<dim3(BB, 4), 256, 0, stream>>>(Sh, t1hT, xf,      t2h, nullptr, 1);
  k_gemm_w<<<BB * NN / 128, 256, 0, stream>>>(xh, t1h, t2h, w1hT, cheb_b1,
                                              y1f, y1h, y1hT, DD, 0);

  // ---- layer 2 ----
  k_gemm_sx<<<dim3(BB, 4), 256, 0, stream>>>(Sh, y1hT, nullptr, t1h, t1hT, 0);
  k_gemm_sx<<<dim3(BB, 4), 256, 0, stream>>>(Sh, t1hT, y1f,     t2h, nullptr, 1);
  k_gemm_w<<<BB * NN / 128, 256, 0, stream>>>(y1h, t1h, t2h, w2hT, cheb_b2,
                                              out, nullptr, nullptr, OUTC, 1);
}
